// MILAttention_30064771072754
// MI455X (gfx1250) — matrix operand order repaired and checked by reading, not compiled, not run on hardware
//
#include <hip/hip_runtime.h>

#define BZ 8
#define PZ 8192
#define FZ 768
#define FI 256

#define NT (FI / 16)   // 16 column tiles
#define KT (FZ / 32)   // 24 K tiles

typedef __attribute__((ext_vector_type(16))) __bf16 v16bf;
typedef __attribute__((ext_vector_type(8)))  float  v8f;

// Packed weight layout: fragment (nt, kt), lane l -> 16 bf16 (32 bytes) at
//   P + ((nt*KT + kt)*32 + l) * 16
// where element j = W[nt*16 + (l&15)][kt*32 + 16*(l>>4) + j]
// (B operand in this file's contiguous k order: lane holds N=l%16, K run = 16h..16h+15)

// ---------------------------------------------------------------------------
// One-shot pre-pack of V_w and U_w (f32 row-major [FI][FZ]) into bf16
// B-fragment order. 2 * NT*KT*32 = 24576 threads, each writes 32 bytes.
// ---------------------------------------------------------------------------
__global__ __launch_bounds__(256) void pack_weights(
    const float* __restrict__ Vw, const float* __restrict__ Uw,
    __bf16* __restrict__ Vp, __bf16* __restrict__ Up) {
  const int t = blockIdx.x * 256 + threadIdx.x;
  const int per_mat = NT * KT * 32;
  const int m    = t / per_mat;
  const int r    = t % per_mat;
  const int nt   = r / (KT * 32);
  const int kt   = (r / 32) % KT;
  const int lane = r & 31;
  const int nl   = lane & 15;
  const int h    = lane >> 4;

  const float* W = m ? Uw : Vw;
  __bf16*      P = m ? Up : Vp;

  const float4* src = reinterpret_cast<const float4*>(
      W + (size_t)(nt * 16 + nl) * FZ + kt * 32 + 16 * h);
  float4 a0 = src[0], a1 = src[1], a2 = src[2], a3 = src[3];
  float f[16] = {a0.x, a0.y, a0.z, a0.w, a1.x, a1.y, a1.z, a1.w,
                 a2.x, a2.y, a2.z, a2.w, a3.x, a3.y, a3.z, a3.w};

  union { v16bf v; float4 f4[2]; } u;
#pragma unroll
  for (int j = 0; j < 16; ++j) u.v[j] = (__bf16)f[j];

  float4* dst = reinterpret_cast<float4*>(
      P + ((size_t)((nt * KT + kt) * 32 + lane)) * 16);
  dst[0] = u.f4[0];
  dst[1] = u.f4[1];
}

// ---------------------------------------------------------------------------
// A-matrix fragment (16x32 bf16, contiguous k order, same as packed B), converted from f32 x:
//   lane L: M = L%16, h = L/16
//   element j:  K = k0 + 16h + j
// ---------------------------------------------------------------------------
__device__ __forceinline__ v16bf load_a_bf16(const float* __restrict__ row,
                                             int k0, int h) {
  const float4* p1 = reinterpret_cast<const float4*>(row + k0 + 16 * h);
  const float4* p2 = reinterpret_cast<const float4*>(row + k0 + 16 * h + 8);
  float4 a0 = p1[0], a1 = p1[1];
  float4 b0 = p2[0], b1 = p2[1];
  float f[16] = {a0.x, a0.y, a0.z, a0.w, a1.x, a1.y, a1.z, a1.w,
                 b0.x, b0.y, b0.z, b0.w, b1.x, b1.y, b1.z, b1.w};
  v16bf r;
#pragma unroll
  for (int j = 0; j < 16; ++j) r[j] = (__bf16)f[j];
  return r;
}

// Pre-packed B fragment: 32 contiguous bytes per lane -> 2x global_load_b128.
__device__ __forceinline__ v16bf load_b_packed(const __bf16* __restrict__ P,
                                               int frag, int lane) {
  const float4* p =
      reinterpret_cast<const float4*>(P + ((size_t)frag * 32 + lane) * 16);
  union { v16bf v; float4 f4[2]; } u;
  u.f4[0] = p[0];
  u.f4[1] = p[1];
  return u.v;
}

__device__ __forceinline__ void sched_fence() {
#if defined(__has_builtin)
#if __has_builtin(__builtin_amdgcn_sched_barrier)
  __builtin_amdgcn_sched_barrier(0);
#endif
#endif
}

// ---------------------------------------------------------------------------
// Logits: att[row] = sum_i sigmoid(x·U^T+Ub)_i * tanh(x·V^T+Vb)_i * w_i + wb
// Grid: 4096 blocks (one 16-row tile), 128 threads = 4 waves.
// Wave w owns n-tiles {4w..4w+3}; 8 independent WMMA accumulator chains,
// register double-buffered prefetch of the next K-step's A + 8 B fragments.
// ---------------------------------------------------------------------------
__global__ __launch_bounds__(128) void mil_logits(
    const float* __restrict__ x,
    const __bf16* __restrict__ Vp, const __bf16* __restrict__ Up,
    const float* __restrict__ Vb,  const float* __restrict__ Ub,
    const float* __restrict__ ww,  const float* __restrict__ wb,
    float* __restrict__ att) {
  const int tile = blockIdx.x;
  const int lane = threadIdx.x & 31;
  const int wave = threadIdx.x >> 5;   // 0..3
  const int h    = lane >> 4;
  const int nl   = lane & 15;

  const float* xrow = x + (size_t)(tile * 16 + nl) * FZ;

  int fragbase[4];
  float bias_v[4], bias_u[4], wcol[4];
#pragma unroll
  for (int i = 0; i < 4; ++i) {
    const int nt  = wave * 4 + i;
    const int col = nt * 16 + nl;
    fragbase[i] = nt * KT;
    bias_v[i] = Vb[col];
    bias_u[i] = Ub[col];
    wcol[i]   = ww[col];
  }

  v8f accv[4] = {}, accu[4] = {};

  // double buffers (ping-pong, fully unrolled so indices are static)
  v16bf a_buf[2];
  v16bf bvb[2][4], bub[2][4];

  // prologue: fill buffer 0 with K-step 0
  a_buf[0] = load_a_bf16(xrow, 0, h);
#pragma unroll
  for (int i = 0; i < 4; ++i) {
    bvb[0][i] = load_b_packed(Vp, fragbase[i], lane);
    bub[0][i] = load_b_packed(Up, fragbase[i], lane);
  }

#pragma unroll
  for (int kt = 0; kt < KT; ++kt) {
    const int cur = kt & 1;
    const int nxt = cur ^ 1;

    // prefetch K-step kt+1 into the other buffer (issued before the WMMAs)
    if (kt + 1 < KT) {
      a_buf[nxt] = load_a_bf16(xrow, (kt + 1) * 32, h);
#pragma unroll
      for (int i = 0; i < 4; ++i) {
        bvb[nxt][i] = load_b_packed(Vp, fragbase[i] + kt + 1, lane);
        bub[nxt][i] = load_b_packed(Up, fragbase[i] + kt + 1, lane);
      }
    }
    sched_fence();  // keep the prefetch clause above the WMMA block

#pragma unroll
    for (int i = 0; i < 4; ++i) {
      accv[i] = __builtin_amdgcn_wmma_f32_16x16x32_bf16(
          false, a_buf[cur], false, bvb[cur][i], (short)0, accv[i], false, false);
      accu[i] = __builtin_amdgcn_wmma_f32_16x16x32_bf16(
          false, a_buf[cur], false, bub[cur][i], (short)0, accu[i], false, false);
    }
    sched_fence();  // keep this WMMA block ahead of the next prefetch
  }

  // gated combine + per-lane partial sum over this wave's 64 columns
  float pacc[8];
#pragma unroll
  for (int r = 0; r < 8; ++r) pacc[r] = 0.0f;
#pragma unroll
  for (int i = 0; i < 4; ++i) {
#pragma unroll
    for (int r = 0; r < 8; ++r) {
      const float v = tanhf(accv[i][r] + bias_v[i]);
      const float u = 1.0f / (1.0f + __expf(-(accu[i][r] + bias_u[i])));
      pacc[r] += u * v * wcol[i];
    }
  }

  // reduce over N within each 16-lane group (C layout: N = lane%16)
#pragma unroll
  for (int r = 0; r < 8; ++r) {
#pragma unroll
    for (int off = 1; off < 16; off <<= 1)
      pacc[r] += __shfl_xor(pacc[r], off, 32);
  }

  // C layout: VGPR r -> M=r (lanes 0-15) / M=r+8 (lanes 16-31)
  __shared__ float part[4][16];
  if (nl == 0) {
#pragma unroll
    for (int r = 0; r < 8; ++r) part[wave][h * 8 + r] = pacc[r];
  }
  __syncthreads();

  if (threadIdx.x < 16) {
    const float s = part[0][threadIdx.x] + part[1][threadIdx.x] +
                    part[2][threadIdx.x] + part[3][threadIdx.x] + wb[0];
    att[(size_t)tile * 16 + threadIdx.x] = s;
  }
}

// ---------------------------------------------------------------------------
// Masked softmax per bag. Grid: BZ blocks x 256 threads (8 wave32 waves).
// ---------------------------------------------------------------------------
__global__ __launch_bounds__(256) void mil_softmax(
    const float* __restrict__ att, const int* __restrict__ nonpad,
    float* __restrict__ out) {
  const int b = blockIdx.x;
  const int n = nonpad[b];
  const float* a = att + (size_t)b * PZ;
  float*       o = out + (size_t)b * PZ;
  const int t = threadIdx.x;

  __shared__ float smax[8], ssum[8];

  float lmax = -3.402823466e38f;
  for (int p = t; p < PZ; p += 256)
    if (p < n) lmax = fmaxf(lmax, a[p]);
#pragma unroll
  for (int off = 16; off > 0; off >>= 1)
    lmax = fmaxf(lmax, __shfl_xor(lmax, off, 32));
  if ((t & 31) == 0) smax[t >> 5] = lmax;
  __syncthreads();

  float bmax = smax[0];
#pragma unroll
  for (int i = 1; i < 8; ++i) bmax = fmaxf(bmax, smax[i]);

  float lsum = 0.0f;
  for (int p = t; p < PZ; p += 256)
    if (p < n) lsum += __expf(a[p] - bmax);
#pragma unroll
  for (int off = 16; off > 0; off >>= 1)
    lsum += __shfl_xor(lsum, off, 32);
  if ((t & 31) == 0) ssum[t >> 5] = lsum;
  __syncthreads();

  float tot = 0.0f;
#pragma unroll
  for (int i = 0; i < 8; ++i) tot += ssum[i];
  const float inv = 1.0f / tot;

  for (int p = t; p < PZ; p += 256)
    o[p] = (p < n) ? __expf(a[p] - bmax) * inv : 0.0f;
}

// ---------------------------------------------------------------------------
extern "C" void kernel_launch(void* const* d_in, const int* in_sizes, int n_in,
                              void* d_out, int out_size, void* d_ws,
                              size_t ws_size, hipStream_t stream) {
  const float* x      = (const float*)d_in[0];
  const int*   nonpad = (const int*)  d_in[1];
  const float* Vw     = (const float*)d_in[2];
  const float* Vb     = (const float*)d_in[3];
  const float* Uw     = (const float*)d_in[4];
  const float* Ub     = (const float*)d_in[5];
  const float* ww     = (const float*)d_in[6];
  const float* wb     = (const float*)d_in[7];

  // workspace layout: [att: BZ*PZ f32][Vp: packed bf16][Up: packed bf16]
  char* ws = (char*)d_ws;
  float*  att = (float*)ws;                                   // 256 KB
  __bf16* Vp  = (__bf16*)(ws + (size_t)BZ * PZ * sizeof(float));
  __bf16* Up  = Vp + (size_t)NT * KT * 32 * 16;               // +384 KB each
  float* out = (float*)d_out;

  const int pack_threads = 2 * NT * KT * 32;                  // 24576
  pack_weights<<<pack_threads / 256, 256, 0, stream>>>(Vw, Uw, Vp, Up);

  const int tiles = (BZ * PZ) / 16;                           // 4096
  mil_logits<<<tiles, 128, 0, stream>>>(x, Vp, Up, Vb, Ub, ww, wb, att);
  mil_softmax<<<BZ, 256, 0, stream>>>(att, nonpad, out);
}
